// RegularGrid_50311246905919
// MI455X (gfx1250) — matrix operand order repaired and checked by reading, not compiled, not run on hardware
//
#include <hip/hip_runtime.h>
#include <hip/hip_bf16.h>
#include <math.h>

typedef __attribute__((ext_vector_type(16))) _Float16 v16h;
typedef __attribute__((ext_vector_type(2)))  _Float16 v2h;
typedef __attribute__((ext_vector_type(8)))  float    v8f;

#define N_RAYS     2048
#define GRID_R     160
#define GRID_VOL   (160 * 160 * 160)
#define N_SAMPLES  480
#define N_PTS      479          /* pts uses inter[:, :-1] */
#define N_CHUNKS   15           /* 15 * 32 = 480 lane-slots */
#define WAVES_PER_BLOCK 8

#define AABB_MIN_F   (-1.3f)
#define AABB_SIZE_F  (2.6f)
#define NEAR_F       (0.2f)
#define FAR_F        (6.0f)

union f16x2_u { v2h h2; float f; unsigned u; };

__device__ __forceinline__ float fast_sigmoid(float x) {
    // v_exp_f32 + v_rcp_f32 instead of libm expf + precise division
    return __builtin_amdgcn_rcpf(1.0f + __expf(-x));
}

__global__ __launch_bounds__(256)
void plenoxel_render_kernel(const float* __restrict__ rays_o,
                            const float* __restrict__ rays_d,
                            const float* __restrict__ sigma_grid,
                            const float* __restrict__ rgb_grid,
                            float* __restrict__ out)
{
    const int lane = threadIdx.x & 31;
    const int wave = threadIdx.x >> 5;
    const int ray  = blockIdx.x * WAVES_PER_BLOCK + wave;   // grid sized exactly

    const float ox = rays_o[ray * 3 + 0];
    const float oy = rays_o[ray * 3 + 1];
    const float oz = rays_o[ray * 3 + 2];
    const float dx = rays_d[ray * 3 + 0];
    const float dy = rays_d[ray * 3 + 1];
    const float dz = rays_d[ray * 3 + 2];

    // ---- ray march setup (matches get_intersections) ----
    const float step = AABB_SIZE_F * 1.7320508075688772f / (float)N_SAMPLES;
    const float idx_ = 1.0f / dx, idy_ = 1.0f / dy, idz_ = 1.0f / dz;
    float tx = fminf(( 1.3f - ox) * idx_, (-1.3f - ox) * idx_);
    float ty = fminf(( 1.3f - oy) * idy_, (-1.3f - oy) * idy_);
    float tz = fminf(( 1.3f - oz) * idz_, (-1.3f - oz) * idz_);
    float tstart = fmaxf(fmaxf(tx, ty), tz);
    tstart = fminf(fmaxf(tstart, NEAR_F), FAR_F);

    const float dlen = sqrtf(dx * dx + dy * dy + dz * dz);
    const float dist = step * dlen;                  // uniform per reference

    // ---- SH basis of the (normalized) direction ----
    const float inl = 1.0f / dlen;
    const float nx = dx * inl, ny = dy * inl, nz = dz * inl;
    const float C0 = 0.28209479177387814f;
    const float C1 = 0.4886025119029199f;
    float sh0 = C0;
    float sh1 = -C1 * ny;
    float sh2 =  C1 * nz;
    float sh3 = -C1 * nx;
    float sh4 =  1.0925484305920792f  * nx * ny;
    float sh5 = -1.0925484305920792f  * ny * nz;
    float sh6 =  0.31539156525252005f * (2.0f * nz * nz - nx * nx - ny * ny);
    float sh7 = -1.0925484305920792f  * nx * nz;
    float sh8 =  0.5462742152960396f  * (nx * nx - ny * ny);

    // Stage SH coefficients in LDS so the B-matrix build can index dynamically.
    __shared__ float sh_lds[WAVES_PER_BLOCK][16];
    if (lane == 0) {
        sh_lds[wave][0] = sh0; sh_lds[wave][1] = sh1; sh_lds[wave][2] = sh2;
        sh_lds[wave][3] = sh3; sh_lds[wave][4] = sh4; sh_lds[wave][5] = sh5;
        sh_lds[wave][6] = sh6; sh_lds[wave][7] = sh7; sh_lds[wave][8] = sh8;
    }
    __syncthreads();

    // ---- Build B matrix [32 x 16] f16, block-diagonal SH (per ISA B layout:
    //      lanes 0-15 hold K=0..7,16..23, lanes 16-31 hold K=8..15,24..31, N = lane%16) ----
    const int ncol = lane & 15;
    v16h Bm;
#pragma unroll
    for (int e = 0; e < 16; ++e) {
        const int k = ((e < 8) ? e : (e + 8)) + ((lane & 16) >> 1);          // K index
        float bv = 0.0f;
        const int j = k - 9 * ncol;             // coefficient within channel
        if (ncol < 3 && j >= 0 && j < 9) bv = sh_lds[wave][j];
        Bm[e] = (_Float16)bv;
    }

    const float chan_mask = (ncol < 3) ? 1.0f : 0.0f;  // branchless D readback
    const bool  hi_half   = (lane >= 16);

    const float scale = (float)(GRID_R - 1) / AABB_SIZE_F;   // world -> voxel

    float T = 1.0f;          // running transmittance across chunks
    float absl_sum = 0.0f;   // sum of abs_light (for white background)
    float racc = 0.0f;       // rgb accumulator for channel (lane%16), if < 3

    for (int chunk = 0; chunk < N_CHUNKS; ++chunk) {
        const int  s      = chunk * 32 + lane;
        const bool active = (s < N_PTS);
        const float t = tstart + (float)s * step;
        const float px = ox + dx * t;
        const float py = oy + dy * t;
        const float pz = oz + dz * t;

        // grid_sample convention: coord x -> Z (fastest) dim, y -> Y, z -> X
        const float fx = (px - AABB_MIN_F) * scale;
        const float fy = (py - AABB_MIN_F) * scale;
        const float fz = (pz - AABB_MIN_F) * scale;

        float feat[27];
#pragma unroll
        for (int c = 0; c < 27; ++c) feat[c] = 0.0f;
        float sig = 0.0f;

        if (fx > -1.0f && fx < (float)GRID_R &&
            fy > -1.0f && fy < (float)GRID_R &&
            fz > -1.0f && fz < (float)GRID_R) {
            const float x0f = floorf(fx), y0f = floorf(fy), z0f = floorf(fz);
            const float wx = fx - x0f, wy = fy - y0f, wz = fz - z0f;
            const int x0 = (int)x0f, y0 = (int)y0f, z0 = (int)z0f;
#pragma unroll
            for (int cz = 0; cz < 2; ++cz) {
#pragma unroll
                for (int cy = 0; cy < 2; ++cy) {
#pragma unroll
                    for (int cx = 0; cx < 2; ++cx) {
                        const int ix = x0 + cx, iy = y0 + cy, iz = z0 + cz;
                        const bool ok = ((unsigned)ix < (unsigned)GRID_R) &&
                                        ((unsigned)iy < (unsigned)GRID_R) &&
                                        ((unsigned)iz < (unsigned)GRID_R);
                        if (ok) {
                            const float w = (cx ? wx : 1.0f - wx) *
                                            (cy ? wy : 1.0f - wy) *
                                            (cz ? wz : 1.0f - wz);
                            const int lin = (iz * GRID_R + iy) * GRID_R + ix;
                            sig += w * sigma_grid[lin];
#pragma unroll
                            for (int c = 0; c < 27; ++c)
                                feat[c] += w * rgb_grid[c * GRID_VOL + lin];
                        }
                    }
                }
            }
        }

        const float sigma = fmaxf(sig, 0.0f);
        const float alpha = active ? (1.0f - __expf(-sigma * dist)) : 0.0f;
        const float f     = active ? (1.0f - alpha + 1e-10f) : 1.0f;

        // ---- wave32 multiplicative scan for cum_light ----
        float incl = f;
#pragma unroll
        for (int off = 1; off < 32; off <<= 1) {
            const float p = __shfl_up(incl, off, 32);
            if (lane >= off) incl *= p;
        }
        float excl = __shfl_up(incl, 1, 32);
        if (lane == 0) excl = 1.0f;
        const float absl = alpha * (T * excl);
        absl_sum += absl;
        T *= __shfl(incl, 31, 32);

        // ---- pack this lane's A-row contribution once: (feat[k0], feat[k0+8])
        //      as f16x2, so each WMMA block needs only ONE bpermute/element ----
        float pk[16];
#pragma unroll
        for (int e = 0; e < 16; ++e) {
            const int k0 = (e < 8) ? e : (e + 8);
            f16x2_u u;
            u.h2[0] = (_Float16)feat[k0];
            u.h2[1] = (_Float16)((k0 + 8 < 27) ? feat[k0 + 8] : 0.0f);
            pk[e] = u.f;
        }

        // ---- SH contraction as WMMA: D[16x16] = A[16x32] x B[32x16] ----
        // Two 16-sample blocks per 32-sample chunk.
#pragma unroll
        for (int b = 0; b < 2; ++b) {
            const int src = (lane & 15) + (b << 4);   // lane owning row m
            v16h Am;
#pragma unroll
            for (int e = 0; e < 16; ++e) {
                f16x2_u g;
                g.f = __shfl(pk[e], src, 32);
                // lanes 0-15 take K=k0 (lo half), lanes 16-31 take K=k0+8 (hi)
                Am[e] = hi_half ? g.h2[1] : g.h2[0];
            }
            v8f cz = {};
            v8f Dm = __builtin_amdgcn_wmma_f32_16x16x32_f16(
                false, Am, false, Bm, (short)0, cz, false, false);

#pragma unroll
            for (int v = 0; v < 8; ++v) {
                const int m = v + ((lane >> 4) << 3);  // D row for this lane half
                const float w = __shfl(absl, m + (b << 4), 32);
                racc += (w * chan_mask) * fast_sigmoid(Dm[v]);
            }
        }
    }

    // ---- reductions and output ----
    float tot = absl_sum;
#pragma unroll
    for (int off = 16; off >= 1; off >>= 1)
        tot += __shfl_xor(tot, off, 32);

    racc += __shfl_xor(racc, 16, 32);   // combine M=0..7 and M=8..15 halves
    if (lane < 3)
        out[ray * 3 + lane] = racc + (1.0f - tot);    // WHITE_BKGD
}

extern "C" void kernel_launch(void* const* d_in, const int* in_sizes, int n_in,
                              void* d_out, int out_size, void* d_ws, size_t ws_size,
                              hipStream_t stream) {
    (void)in_sizes; (void)n_in; (void)d_ws; (void)ws_size; (void)out_size;
    const float* rays_o     = (const float*)d_in[0];
    const float* rays_d     = (const float*)d_in[1];
    const float* sigma_grid = (const float*)d_in[2];
    const float* rgb_grid   = (const float*)d_in[3];
    float* out = (float*)d_out;

    dim3 block(32 * WAVES_PER_BLOCK);
    dim3 grid(N_RAYS / WAVES_PER_BLOCK);
    plenoxel_render_kernel<<<grid, block, 0, stream>>>(
        rays_o, rays_d, sigma_grid, rgb_grid, out);
}
